// SparseTransformerLayer_17317308138046
// MI455X (gfx1250) — matrix operand order repaired and checked by reading, not compiled, not run on hardware
//
#include <hip/hip_runtime.h>
#include <hip/hip_bf16.h>

// ---------------------------------------------------------------------------
// CDNA5 (gfx1250) transformer layer:
//   LN1 -> QKV proj (WMMA, head-split epilogue) ->
//   fused [scores WMMA -> entmax1.5 -> P@V WMMA] (scores live in LDS) ->
//   Wo proj + residual -> LN2 -> FFN1(ReLU) -> FFN2 + residual
// All GEMMs via v_wmma_f32_16x16x32_bf16 (wave32, 16x16 tiles, K=32 steps).
// ---------------------------------------------------------------------------

typedef __bf16 bf16_t;
typedef __attribute__((ext_vector_type(16))) __bf16 v16bf;
typedef __attribute__((ext_vector_type(8)))  __bf16 v8bf;
typedef __attribute__((ext_vector_type(8)))  float  v8f;

union V16U { v16bf v; v8bf h[2]; };

static constexpr int TB   = 2;     // batch
static constexpr int TS   = 1024;  // sequence
static constexpr int TD   = 1024;  // d_model
static constexpr int THD  = 16;    // heads
static constexpr int TDK  = 64;    // head dim
static constexpr int TDFF = 4096;  // ffn dim

static constexpr int QT   = 32;        // queries per attention block
static constexpr int LDSW = TS + 8;    // LDS score row stride (f32)

// ---------------------------------------------------------------------------
// WMMA fragment loaders (layouts per CDNA5 ISA 7.12.2, wave32)
// A (16x32 bf16, row-major [M,K]):
//   lane: M = lane&15, half = lane>>4
//   elems 0..7  = A[M][k0 + half*8 + 0..7]
//   elems 8..15 = A[M][k0 + half*8 + 16..23]
// B (32x16 bf16, row-major [K,N]):
//   lane: N = lane&15, half = lane>>4
//   elems 0..7  = B[k0 + half*8 + 0..7][N]
//   elems 8..15 = B[k0 + half*8 + 16..23][N]
// C/D (16x16 f32): lane col N = lane&15, VGPR i -> row M = i + 8*(lane>>4)
// ---------------------------------------------------------------------------
__device__ inline v16bf load_a_frag(const bf16_t* A, int lda, int m0, int k0, int lane) {
    int row = m0 + (lane & 15);
    int kb  = k0 + ((lane >> 4) << 3);
    const bf16_t* p = A + (long long)row * lda + kb;
    V16U u;
    u.h[0] = *(const v8bf*)(p);
    u.h[1] = *(const v8bf*)(p + 16);
    return u.v;
}

__device__ inline v16bf load_b_frag(const bf16_t* B, int ldb, int k0, int n0, int lane) {
    const int col = n0 + (lane & 15);
    const int kb  = k0 + ((lane >> 4) << 3);
    const bf16_t* p = B + (long long)kb * ldb + col;
    v16bf v;
#pragma unroll
    for (int e = 0; e < 8; ++e) v[e]     = p[(long long)e * ldb];
#pragma unroll
    for (int e = 0; e < 8; ++e) v[8 + e] = p[(long long)(16 + e) * ldb];
    return v;
}

__device__ inline v8f wmma_bf16(v16bf a, v16bf b, v8f c) {
    return __builtin_amdgcn_wmma_f32_16x16x32_bf16(false, a, false, b, (short)0, c,
                                                   false, false);
}

// ---------------------------------------------------------------------------
// Generic batched GEMM: C = alpha * A@B + bias (+relu) (+resid)
// batch z: offsets = (z/H)*s?0 + (z%H)*s?1  (element offsets)
// block = 128 threads (4 waves); wave w -> rows [by*64 + w*16, +16),
// cols [bx*64, +64) as four 16x16 WMMA accumulators sharing one A fragment.
// Requires M%64==0, N%64==0, K%32==0.
// Epilogue bf16 index modes:
//   0: normal row-major (cOff + row*ldc + col)
//   1: head-split [b,h,s,d]     (row = b*S+s, col = h*64+d)   -- q, v
//   2: head-split T [b,h,d,s]   (k^T: scores B-operand layout)
// ---------------------------------------------------------------------------
struct GemmP {
    const bf16_t* A; long long sA0, sA1; int lda;
    const bf16_t* B; long long sB0, sB1; int ldb;
    float* Cf; bf16_t* Cb; long long sC0, sC1; int ldc;
    const float* bias;   // length N (indexed by absolute column) or null
    const float* resid;  // normal-mode indexing, or null
    float alpha; int relu; int mode;
    int M, N, K, H;
};

__device__ inline long long c_index(int mode, long long cOff, int row, int col, int ldc) {
    if (mode != 0) {
        const int b = row >> 10;            // row / TS
        const int s = row & (TS - 1);
        const int h = col >> 6;             // col / TDK
        const int d = col & (TDK - 1);
        if (mode == 1) return ((long long)(b * THD + h) * TS + s) * TDK + d;
        return ((long long)(b * THD + h) * TDK + d) * TS + s;  // mode 2
    }
    return cOff + (long long)row * ldc + col;
}

__global__ __launch_bounds__(128) void gemm_wmma_kernel(GemmP p) {
    const int lane = threadIdx.x & 31;
    const int wave = threadIdx.x >> 5;
    const int z    = blockIdx.z;
    const int zo   = z / p.H;
    const int zi   = z % p.H;

    const bf16_t* A = p.A + zo * p.sA0 + zi * p.sA1;
    const bf16_t* B = p.B + zo * p.sB0 + zi * p.sB1;
    const long long cOff = zo * p.sC0 + zi * p.sC1;

    const int m0 = blockIdx.y * 64 + wave * 16;
    const int n0 = blockIdx.x * 64;

    v8f acc0 = {}, acc1 = {}, acc2 = {}, acc3 = {};

    for (int kk = 0; kk < p.K; kk += 32) {
        v16bf a  = load_a_frag(A, p.lda, m0, kk, lane);
        v16bf b0 = load_b_frag(B, p.ldb, kk, n0 +  0, lane);
        v16bf b1 = load_b_frag(B, p.ldb, kk, n0 + 16, lane);
        v16bf b2 = load_b_frag(B, p.ldb, kk, n0 + 32, lane);
        v16bf b3 = load_b_frag(B, p.ldb, kk, n0 + 48, lane);
        acc0 = wmma_bf16(a, b0, acc0);
        acc1 = wmma_bf16(a, b1, acc1);
        acc2 = wmma_bf16(a, b2, acc2);
        acc3 = wmma_bf16(a, b3, acc3);
    }

    const int half  = lane >> 4;
    const int col16 = lane & 15;

    v8f accs[4] = {acc0, acc1, acc2, acc3};
#pragma unroll
    for (int j = 0; j < 4; ++j) {
        const int col = n0 + j * 16 + col16;
        const float bia = p.bias ? p.bias[col] : 0.0f;
#pragma unroll
        for (int i = 0; i < 8; ++i) {
            const int row = m0 + half * 8 + i;
            const long long nidx = cOff + (long long)row * p.ldc + col;
            float v = accs[j][i] * p.alpha + bia;
            if (p.relu)  v = v > 0.0f ? v : 0.0f;
            if (p.resid) v += p.resid[nidx];
            if (p.Cf) p.Cf[nidx] = v;
            if (p.Cb) p.Cb[c_index(p.mode, cOff, row, col, p.ldc)] = (bf16_t)v;
        }
    }
}

// ---------------------------------------------------------------------------
// Fused attention: per (b,h, 32-query tile):
//   phase 1: scores = q @ k^T into LDS (f32)
//   phase 2: entmax-1.5 per row (bisection, wave-local, __shfl_xor reductions)
//   phase 3: ctx = P @ v, heads merged into [B,S,D] bf16 output
// block = 128 threads (4 waves). Dynamic LDS: QT * LDSW * 4 = 132KB.
// ---------------------------------------------------------------------------
__global__ __launch_bounds__(128) void attn_fused_kernel(
        const bf16_t* __restrict__ qh,   // [B*H, S, 64]
        const bf16_t* __restrict__ kT,   // [B*H, 64, S]
        const bf16_t* __restrict__ vh,   // [B*H, S, 64]
        const unsigned char* __restrict__ mask,  // [B, S, S]
        bf16_t* __restrict__ ctxm) {     // [B, S, D]
    extern __shared__ float sc[];        // [QT][LDSW]

    const int lane = threadIdx.x & 31;
    const int wave = threadIdx.x >> 5;
    const int bh   = blockIdx.y;         // b*H + h
    const int b    = bh >> 4;
    const int h    = bh & (THD - 1);
    const int q0   = blockIdx.x * QT;

    const bf16_t* Q = qh + (long long)bh * TS * TDK;
    const bf16_t* K = kT + (long long)bh * TDK * TS;
    const bf16_t* V = vh + (long long)bh * TS * TDK;

    // ---- phase 1: scores tile (32 q-rows x 1024 keys) ----
    {
        const int rowTile = (wave & 1) * 16;       // 0 or 16 within tile
        const int keyBase = (wave >> 1) * 512;     // 0 or 512
        const int m0      = q0 + rowTile;

        const v16bf a0 = load_a_frag(Q, TDK, m0, 0,  lane);
        const v16bf a1 = load_a_frag(Q, TDK, m0, 32, lane);

        const int half  = lane >> 4;
        const int col16 = lane & 15;

        for (int n = 0; n < 512; n += 16) {
            const int n0 = keyBase + n;
            v16bf b0 = load_b_frag(K, TS, 0,  n0, lane);
            v16bf b1 = load_b_frag(K, TS, 32, n0, lane);
            v8f acc = {};
            acc = wmma_bf16(a0, b0, acc);
            acc = wmma_bf16(a1, b1, acc);
#pragma unroll
            for (int i = 0; i < 8; ++i) {
                sc[(rowTile + half * 8 + i) * LDSW + n0 + col16] = acc[i];
            }
        }
    }
    __syncthreads();

    // ---- phase 2: entmax-1.5 per row (wave handles 8 rows) ----
    for (int rr = 0; rr < 8; ++rr) {
        const int lrow = wave * 8 + rr;
        const int q    = q0 + lrow;
        float* row = &sc[lrow * LDSW];
        const unsigned char* mrow = mask + ((long long)b * TS + q) * (long long)TS;

        float z[32];
        float lmax = -3.0e38f;
#pragma unroll
        for (int j = 0; j < 32; ++j) {
            const int idx = lane + j * 32;
            // fold in 1/sqrt(dk) (=1/8) and the entmax x/2 => *0.0625
            float s = mrow[idx] ? row[idx] * 0.0625f : -1.0e9f;
            z[j] = s;
            lmax = fmaxf(lmax, s);
        }
#pragma unroll
        for (int o = 16; o > 0; o >>= 1) lmax = fmaxf(lmax, __shfl_xor(lmax, o, 32));
#pragma unroll
        for (int j = 0; j < 32; ++j) z[j] -= lmax;

        // f(tau) = sum (z - tau)_+^2 ; f(-1) >= 1 >= f(0); bisection on tau.
        float lo = -1.0f, hi = 0.0f;
        for (int it = 0; it < 30; ++it) {
            const float tau = 0.5f * (lo + hi);
            float ls = 0.0f;
#pragma unroll
            for (int j = 0; j < 32; ++j) {
                float d = z[j] - tau;
                d = d > 0.0f ? d : 0.0f;
                ls += d * d;
            }
#pragma unroll
            for (int o = 16; o > 0; o >>= 1) ls += __shfl_xor(ls, o, 32);
            if (ls >= 1.0f) lo = tau; else hi = tau;
        }
        const float tau = 0.5f * (lo + hi);
#pragma unroll
        for (int j = 0; j < 32; ++j) {
            float d = z[j] - tau;
            d = d > 0.0f ? d : 0.0f;
            row[lane + j * 32] = d * d;   // probabilities back in place
        }
    }
    __syncthreads();

    // ---- phase 3: ctx = P @ V (K = 1024 keys) ----
    {
        const int rowTile = (wave & 1) * 16;
        const int n0      = (wave >> 1) * 32;   // 0 or 32 of the 64 head dims

        v8f acc0 = {}, acc1 = {};
        for (int kk = 0; kk < TS; kk += 32) {
            // A fragment from LDS probabilities (f32 -> bf16)
            const int arow = rowTile + (lane & 15);
            const int kb   = kk + ((lane >> 4) << 3);
            const float* ap = &sc[arow * LDSW + kb];
            v16bf a;
#pragma unroll
            for (int e = 0; e < 8; ++e) a[e]     = (bf16_t)ap[e];
#pragma unroll
            for (int e = 0; e < 8; ++e) a[8 + e] = (bf16_t)ap[16 + e];

            v16bf b0 = load_b_frag(V, TDK, kk, n0,      lane);
            v16bf b1 = load_b_frag(V, TDK, kk, n0 + 16, lane);
            acc0 = wmma_bf16(a, b0, acc0);
            acc1 = wmma_bf16(a, b1, acc1);
        }

        const int half  = lane >> 4;
        const int col16 = lane & 15;
        bf16_t* outBase = ctxm + (long long)b * TS * TD + h * TDK;
        v8f accs[2] = {acc0, acc1};
#pragma unroll
        for (int j = 0; j < 2; ++j) {
            const int col = n0 + j * 16 + col16;
#pragma unroll
            for (int i = 0; i < 8; ++i) {
                const int row = q0 + rowTile + half * 8 + i;
                outBase[(long long)row * TD + col] = (bf16_t)accs[j][i];
            }
        }
    }
}

// ---------------------------------------------------------------------------
// LayerNorm: one 256-thread block per row of length D (=1024), bf16 output.
// ---------------------------------------------------------------------------
__global__ __launch_bounds__(256) void ln_kernel(const float* __restrict__ x,
                                                 const float* __restrict__ g,
                                                 const float* __restrict__ b,
                                                 bf16_t* __restrict__ out, int D) {
    const long long row = blockIdx.x;
    const float* xr = x + row * D;
    const int t = threadIdx.x;

    float lsum = 0.0f, lsq = 0.0f;
    for (int i = t; i < D; i += 256) {
        float v = xr[i];
        lsum += v; lsq += v * v;
    }
    __shared__ float s1[256], s2[256];
    s1[t] = lsum; s2[t] = lsq;
    __syncthreads();
    for (int o = 128; o > 0; o >>= 1) {
        if (t < o) { s1[t] += s1[t + o]; s2[t] += s2[t + o]; }
        __syncthreads();
    }
    const float mean = s1[0] * (1.0f / D);
    const float var  = s2[0] * (1.0f / D) - mean * mean;
    const float rstd = rsqrtf(var + 1e-5f);

    bf16_t* orow = out + row * D;
    for (int i = t; i < D; i += 256) {
        float v = (xr[i] - mean) * rstd * g[i] + b[i];
        orow[i] = (bf16_t)v;
    }
}

// ---------------------------------------------------------------------------
// f32 -> bf16 conversion
// ---------------------------------------------------------------------------
__global__ __launch_bounds__(256) void cvt_kernel(const float* __restrict__ src,
                                                  bf16_t* __restrict__ dst, long long n) {
    const long long i = (long long)blockIdx.x * 256 + threadIdx.x;
    if (i < n) dst[i] = (bf16_t)src[i];
}

// ---------------------------------------------------------------------------
// Host-side plumbing
// ---------------------------------------------------------------------------
static void launch_gemm(hipStream_t st,
                        const bf16_t* A, long long sA0, long long sA1, int lda,
                        const bf16_t* B, long long sB0, long long sB1, int ldb,
                        float* Cf, bf16_t* Cb, long long sC0, long long sC1, int ldc,
                        const float* bias, const float* resid, float alpha, int relu,
                        int mode, int M, int N, int K, int H, int nbatch) {
    GemmP p;
    p.A = A; p.sA0 = sA0; p.sA1 = sA1; p.lda = lda;
    p.B = B; p.sB0 = sB0; p.sB1 = sB1; p.ldb = ldb;
    p.Cf = Cf; p.Cb = Cb; p.sC0 = sC0; p.sC1 = sC1; p.ldc = ldc;
    p.bias = bias; p.resid = resid; p.alpha = alpha; p.relu = relu; p.mode = mode;
    p.M = M; p.N = N; p.K = K; p.H = H;
    dim3 grid(N / 64, M / 64, nbatch);
    dim3 blk(128, 1, 1);
    gemm_wmma_kernel<<<grid, blk, 0, st>>>(p);
}

extern "C" void kernel_launch(void* const* d_in, const int* in_sizes, int n_in,
                              void* d_out, int out_size, void* d_ws, size_t ws_size,
                              hipStream_t stream) {
    (void)in_sizes; (void)n_in; (void)out_size; (void)ws_size;

    const float*         x    = (const float*)d_in[0];
    const unsigned char* mask = (const unsigned char*)d_in[1];
    const float* Wq = (const float*)d_in[2];   const float* bq = (const float*)d_in[3];
    const float* Wk = (const float*)d_in[4];   const float* bk = (const float*)d_in[5];
    const float* Wv = (const float*)d_in[6];   const float* bv = (const float*)d_in[7];
    const float* Wo = (const float*)d_in[8];   const float* bo = (const float*)d_in[9];
    const float* W1 = (const float*)d_in[10];  const float* b1 = (const float*)d_in[11];
    const float* W2 = (const float*)d_in[12];  const float* b2 = (const float*)d_in[13];
    const float* g1 = (const float*)d_in[14];  const float* be1 = (const float*)d_in[15];
    const float* g2 = (const float*)d_in[16];  const float* be2 = (const float*)d_in[17];
    float* out = (float*)d_out;

    const long long BS  = (long long)TB * TS;        // 2048 rows
    const long long DD  = (long long)TD * TD;        // 1M
    const long long DFF = (long long)TD * TDFF;      // 4M
    const long long BSD = BS * TD;                   // 2M

    // workspace carve-up (256B aligned)
    char* w = (char*)d_ws;
    size_t off = 0;
    auto alloc = [&](size_t bytes) -> void* {
        void* p = w + off;
        off = (off + bytes + 255) & ~(size_t)255;
        return p;
    };
    bf16_t* Wq_b  = (bf16_t*)alloc(DD * 2);
    bf16_t* Wk_b  = (bf16_t*)alloc(DD * 2);
    bf16_t* Wv_b  = (bf16_t*)alloc(DD * 2);
    bf16_t* Wo_b  = (bf16_t*)alloc(DD * 2);
    bf16_t* W1_b  = (bf16_t*)alloc(DFF * 2);
    bf16_t* W2_b  = (bf16_t*)alloc(DFF * 2);
    bf16_t* h1_b  = (bf16_t*)alloc(BSD * 2);
    bf16_t* qh    = (bf16_t*)alloc(BSD * 2);   // [B,H,S,64]
    bf16_t* kT    = (bf16_t*)alloc(BSD * 2);   // [B,H,64,S]
    bf16_t* vh    = (bf16_t*)alloc(BSD * 2);   // [B,H,S,64]
    bf16_t* ctxm  = (bf16_t*)alloc(BSD * 2);   // [B,S,D]
    float*  xmid  = (float*) alloc(BSD * 4);
    bf16_t* h2_b  = (bf16_t*)alloc(BSD * 2);
    bf16_t* ff_b  = (bf16_t*)alloc(BS * (long long)TDFF * 2);

    // 1) weight conversions f32 -> bf16
    cvt_kernel<<<(unsigned)((DD  + 255) / 256), 256, 0, stream>>>(Wq, Wq_b, DD);
    cvt_kernel<<<(unsigned)((DD  + 255) / 256), 256, 0, stream>>>(Wk, Wk_b, DD);
    cvt_kernel<<<(unsigned)((DD  + 255) / 256), 256, 0, stream>>>(Wv, Wv_b, DD);
    cvt_kernel<<<(unsigned)((DD  + 255) / 256), 256, 0, stream>>>(Wo, Wo_b, DD);
    cvt_kernel<<<(unsigned)((DFF + 255) / 256), 256, 0, stream>>>(W1, W1_b, DFF);
    cvt_kernel<<<(unsigned)((DFF + 255) / 256), 256, 0, stream>>>(W2, W2_b, DFF);

    // 2) LN1: x -> h1 (bf16)
    ln_kernel<<<(unsigned)BS, 256, 0, stream>>>(x, g1, be1, h1_b, TD);

    // 3) Q/K/V projections with head-split epilogues (no repack pass):
    //    q -> qh [b,h,s,d] (mode 1), k -> kT [b,h,d,s] (mode 2), v -> vh (mode 1)
    launch_gemm(stream, h1_b, 0, 0, TD, Wq_b, 0, 0, TD,
                nullptr, qh, 0, 0, TD, bq, nullptr, 1.0f, 0, 1,
                (int)BS, TD, TD, 1, 1);
    launch_gemm(stream, h1_b, 0, 0, TD, Wk_b, 0, 0, TD,
                nullptr, kT, 0, 0, TD, bk, nullptr, 1.0f, 0, 2,
                (int)BS, TD, TD, 1, 1);
    launch_gemm(stream, h1_b, 0, 0, TD, Wv_b, 0, 0, TD,
                nullptr, vh, 0, 0, TD, bv, nullptr, 1.0f, 0, 1,
                (int)BS, TD, TD, 1, 1);

    // 4) fused attention: scores -> entmax1.5 -> P@V, heads merged into ctxm
    {
        dim3 grid(TS / QT, TB * THD, 1);   // 32 q-tiles x 32 (b,h)
        dim3 blk(128, 1, 1);
        const size_t ldsBytes = (size_t)QT * LDSW * sizeof(float);  // 132KB
        attn_fused_kernel<<<grid, blk, ldsBytes, stream>>>(qh, kT, vh, mask, ctxm);
    }

    // 5) attn out: xmid = x + ctx @ Wo + bo
    launch_gemm(stream, ctxm, 0, 0, TD, Wo_b, 0, 0, TD,
                xmid, nullptr, 0, 0, TD, bo, x, 1.0f, 0, 0,
                (int)BS, TD, TD, 1, 1);

    // 6) LN2: xmid -> h2 (bf16)
    ln_kernel<<<(unsigned)BS, 256, 0, stream>>>(xmid, g2, be2, h2_b, TD);

    // 7) FFN1: relu(h2 @ W1 + b1) -> bf16 [2048, 4096]
    launch_gemm(stream, h2_b, 0, 0, TD, W1_b, 0, 0, TDFF,
                nullptr, ff_b, 0, 0, TDFF, b1, nullptr, 1.0f, 1, 0,
                (int)BS, TDFF, TD, 1, 1);

    // 8) FFN2: out = xmid + ff @ W2 + b2  (final f32 output)
    launch_gemm(stream, ff_b, 0, 0, TDFF, W2_b, 0, 0, TD,
                out, nullptr, 0, 0, TD, b2, xmid, 1.0f, 0, 0,
                (int)BS, TD, TDFF, 1, 1);
}
